// ScaledDotProductAttention_3650722202112
// MI455X (gfx1250) — matrix operand order repaired and checked by reading, not compiled, not run on hardware
//
#include <hip/hip_runtime.h>

#define NB 16
#define NS 2048
#define ND 128
#define SCALE_F 0.08838834764831845f   // 1/sqrt(128)

typedef __attribute__((ext_vector_type(16))) __bf16 v16bf;
typedef __attribute__((ext_vector_type(8)))  __bf16 v8bf;
typedef __attribute__((ext_vector_type(8)))  float  v8f;

__device__ __forceinline__ float rmax16(float x) {
#pragma unroll
  for (int off = 1; off < 16; off <<= 1)
    x = fmaxf(x, __shfl_xor(x, off, 32));
  return x;
}
__device__ __forceinline__ float rsum16(float x) {
#pragma unroll
  for (int off = 1; off < 16; off <<= 1)
    x += __shfl_xor(x, off, 32);
  return x;
}

// A-matrix 16x32 bf16 frag: elems 0..15 = k (16*half)+i
__device__ __forceinline__ v16bf make_a_frag(const float* rowbase, int half) {
  const float* p0 = rowbase + 16 * half;
  const float* p1 = rowbase + 16 * half + 8;
  v16bf a;
#pragma unroll
  for (int i = 0; i < 8; ++i) a[i] = (__bf16)p0[i];
#pragma unroll
  for (int i = 0; i < 8; ++i) a[8 + i] = (__bf16)p1[i];
  return a;
}

// B-matrix 32x16 bf16 frag: lane holds contraction elems (16*half)+0..15 of column n;
// caller passes pointer to the lane's 16 contiguous source floats.
__device__ __forceinline__ v16bf make_b_frag(const float* p) {
  v16bf b;
#pragma unroll
  for (int i = 0; i < 16; ++i) b[i] = (__bf16)p[i];
  return b;
}

__global__ __launch_bounds__(128) void
ScaledDotProductAttention_3650722202112_kernel(const float* __restrict__ Q,
                                               const float* __restrict__ K,
                                               const float* __restrict__ V,
                                               float* __restrict__ O,
                                               float* __restrict__ W) {
  // Per-wave LDS: V^T chunk (d x k, bf16) and P staging tile (m x k, bf16)
  __shared__ __align__(16) __bf16 VT[4][ND][32];  // 32 KB
  __shared__ __align__(16) __bf16 PB[4][16][32];  //  4 KB

  const int tid  = threadIdx.x;
  const int wave = tid >> 5;
  const int lane = tid & 31;
  const int half = lane >> 4;
  const int n16  = lane & 15;

  const int qt = blockIdx.x * 4 + wave;     // global 16-row query tile id
  const int b  = qt >> 7;                   // 128 tiles per batch
  const int qi = qt & 127;
  const int q0 = qi << 4;

  const size_t boff = (size_t)b * NS * ND;
  const float* Qb = Q + boff;
  const float* Kb = K + boff;
  const float* Vb = V + boff;

  // ---- Q A-frags (kept live for both passes) ----
  v16bf qfrag[4];
  {
    const float* qrow = Qb + (size_t)(q0 + n16) * ND;
#pragma unroll
    for (int t = 0; t < 4; ++t) qfrag[t] = make_a_frag(qrow + 32 * t, half);
  }

  v8f acc[8];
#pragma unroll
  for (int dt = 0; dt < 8; ++dt) acc[dt] = v8f{};
  float mrow[8], lrow[8];
#pragma unroll
  for (int r = 0; r < 8; ++r) { mrow[r] = -__builtin_inff(); lrow[r] = 0.0f; }

  const int ntiles = qi + 1;             // 16-key tiles this row-block touches
  const int nchunk = (ntiles + 1) >> 1;  // 32-key chunks (2nd tile may be fully masked)

  // =================== pass 1: flash attention ===================
  for (int jc = 0; jc < nchunk; ++jc) {
    const int k0 = jc * 32;

    // stage V chunk transposed into LDS as bf16: VT[d][k]
    {
      const float* vrow = Vb + (size_t)(k0 + lane) * ND;
#pragma unroll 8
      for (int d = 0; d < ND; ++d) VT[wave][d][lane] = (__bf16)vrow[d];
    }

    // prefetch next chunk's K/V rows
    if (jc + 1 < nchunk) {
      __builtin_prefetch(Kb + (size_t)(k0 + 32 + lane) * ND, 0, 3);
      __builtin_prefetch(Vb + (size_t)(k0 + 32 + lane) * ND, 0, 3);
    }

    // two 16x16 score tiles: S = Q * K^T  (contraction over d in 4 WMMA steps)
    v8f s0 = v8f{}, s1 = v8f{};
    {
      const float* k0row = Kb + (size_t)(k0 + n16) * ND + 16 * half;
      const float* k1row = k0row + 16 * (size_t)ND;
#pragma unroll
      for (int t = 0; t < 4; ++t) {
        v16bf kb = make_b_frag(k0row + 32 * t);
        s0 = __builtin_amdgcn_wmma_f32_16x16x32_bf16(false, qfrag[t], false, kb,
                                                     (short)0, s0, false, false);
      }
#pragma unroll
      for (int t = 0; t < 4; ++t) {
        v16bf kb = make_b_frag(k1row + 32 * t);
        s1 = __builtin_amdgcn_wmma_f32_16x16x32_bf16(false, qfrag[t], false, kb,
                                                     (short)0, s1, false, false);
      }
    }

    // scale + causal mask + online softmax update; stage P (bf16) to LDS
#pragma unroll
    for (int r = 0; r < 8; ++r) {
      const int qrow = q0 + r + 8 * half;
      float a0 = (k0 + n16      <= qrow) ? s0[r] * SCALE_F : -__builtin_inff();
      float a1 = (k0 + 16 + n16 <= qrow) ? s1[r] * SCALE_F : -__builtin_inff();
      float mx = fmaxf(rmax16(fmaxf(a0, a1)), mrow[r]);
      float alpha = __expf(mrow[r] - mx);
      float p0 = __expf(a0 - mx);
      float p1 = __expf(a1 - mx);
      lrow[r] = lrow[r] * alpha + rsum16(p0 + p1);
      mrow[r] = mx;
      PB[wave][r + 8 * half][n16]      = (__bf16)p0;
      PB[wave][r + 8 * half][16 + n16] = (__bf16)p1;
#pragma unroll
      for (int dt = 0; dt < 8; ++dt) acc[dt][r] *= alpha;
    }

    asm volatile("s_wait_dscnt 0" ::: "memory");  // LDS stores visible to cross-lane reads

    // O += P * V  (A = P from LDS in A-layout, B = V^T slices from LDS)
    v16bf pfrag;
    {
      const __bf16* pb = &PB[wave][n16][0];
      v8bf lo = *(const v8bf*)(pb + 16 * half);
      v8bf hi = *(const v8bf*)(pb + 16 * half + 8);
#pragma unroll
      for (int i = 0; i < 8; ++i) { pfrag[i] = lo[i]; pfrag[8 + i] = hi[i]; }
    }
#pragma unroll
    for (int dt = 0; dt < 8; ++dt) {
      const __bf16* vb = &VT[wave][dt * 16 + n16][16 * half];
      v8bf b0 = *(const v8bf*)vb;
      v8bf b1 = *(const v8bf*)(vb + 8);
      v16bf vfrag;
#pragma unroll
      for (int i = 0; i < 8; ++i) { vfrag[i] = b0[i]; vfrag[8 + i] = b1[i]; }
      acc[dt] = __builtin_amdgcn_wmma_f32_16x16x32_bf16(false, pfrag, false, vfrag,
                                                        (short)0, acc[dt], false, false);
    }
    asm volatile("s_wait_dscnt 0" ::: "memory");  // drain reads before next overwrite
  }

  // ---- normalize + write O ----
  float invl[8];
#pragma unroll
  for (int r = 0; r < 8; ++r) invl[r] = 1.0f / lrow[r];
#pragma unroll
  for (int r = 0; r < 8; ++r) {
    float* orow = O + ((size_t)b * NS + q0 + r + 8 * half) * ND;
#pragma unroll
    for (int dt = 0; dt < 8; ++dt) orow[dt * 16 + n16] = acc[dt][r] * invl[r];
  }

  // =================== pass 2: materialize attention weights ===================
  for (int j = 0; j < NS / 16; ++j) {
    const int k0 = j * 16;
    if (j < ntiles) {
      v8f s = v8f{};
      const float* krow = Kb + (size_t)(k0 + n16) * ND + 16 * half;
#pragma unroll
      for (int t = 0; t < 4; ++t) {
        v16bf kb = make_b_frag(krow + 32 * t);
        s = __builtin_amdgcn_wmma_f32_16x16x32_bf16(false, qfrag[t], false, kb,
                                                    (short)0, s, false, false);
      }
#pragma unroll
      for (int r = 0; r < 8; ++r) {
        const int qrow = q0 + r + 8 * half;
        float a = (k0 + n16 <= qrow) ? s[r] * SCALE_F : -__builtin_inff();
        float w = __expf(a - mrow[r]) * invl[r];
        W[((size_t)b * NS + qrow) * NS + k0 + n16] = w;
      }
    } else {
      break;  // rest is the fully-masked region -> vectorized zero fill below
    }
  }
  {
    const float4 z = make_float4(0.f, 0.f, 0.f, 0.f);
#pragma unroll
    for (int r = 0; r < 8; ++r) {
      float* wrow = W + ((size_t)b * NS + q0 + r + 8 * half) * NS;
      for (int c = ntiles * 16 + n16 * 4; c < NS; c += 64)
        *(float4*)(wrow + c) = z;
    }
  }
}

extern "C" void kernel_launch(void* const* d_in, const int* in_sizes, int n_in,
                              void* d_out, int out_size, void* d_ws, size_t ws_size,
                              hipStream_t stream) {
  const float* Q = (const float*)d_in[0];
  const float* K = (const float*)d_in[1];
  const float* V = (const float*)d_in[2];
  float* out = (float*)d_out;                       // [B,S,D] output
  float* wts = out + (size_t)NB * NS * ND;          // [B,S,S] attention weights
  dim3 grid(NB * (NS / 16) / 4);                    // 512 blocks, 4 waves each
  ScaledDotProductAttention_3650722202112_kernel<<<grid, 128, 0, stream>>>(Q, K, V, out, wts);
}
